// ReactionNN_49014166782082
// MI455X (gfx1250) — matrix operand order repaired and checked by reading, not compiled, not run on hardware
//
#include <hip/hip_runtime.h>
#include <hip/hip_bf16.h>
#include <math.h>

// ---------------- problem constants ----------------
#define N_NODES 20000
#define N_EDGES 40000
#define BGRAPH  256
#define DN_     110
#define DE_     8
#define DL_     8
#define DH_     64
#define DHID_   4096
#define STEPS_  4
#define POOL_IT 3
#define H2_     128   // 2*DH
#define HQ_     256   // 2*H2
#define KEF_    17    // 8 edge attrs + 8 rbf + 1 (bias column folding bond_b)

typedef __attribute__((ext_vector_type(16))) _Float16 v16h;
typedef __attribute__((ext_vector_type(8)))  _Float16 v8h;
typedef __attribute__((ext_vector_type(8)))  float    v8f;
typedef _Float16 half_t;

// ---------------- WMMA helpers (CDNA5 gfx1250, wave32) ----------------
__device__ inline v8f wmma_f32_16x16x32(v16h a, v16h b, v8f c) {
  // D = A(16x32 f16) * B(32x16 f16) + C(16x16 f32)
  return __builtin_amdgcn_wmma_f32_16x16x32_f16(
      /*neg_a=*/false, a, /*neg_b=*/false, b,
      /*c_mod=*/(short)0, c, /*reuse_a=*/false, /*reuse_b=*/false);
}

// A fragment: lane l holds row M=m0+(l&15); halves 0..7 = K in {lo..lo+7},
// halves 8..15 = K in {16+lo..16+lo+7}, lo = (l<16)?0:8  (ISA 16-bit A 16x32 table)
__device__ inline v16h load_afrag(const half_t* base, int ld, int m0, int k0) {
  const int l  = threadIdx.x & 31;
  const int M  = m0 + (l & 15);
  const int lo = (l < 16) ? 0 : 8;
  const half_t* p = base + (size_t)M * ld + k0 + lo;
  v8h a0 = *(const v8h*)p;
  v8h a1 = *(const v8h*)(p + 16);
  return __builtin_shufflevector(a0, a1, 0,1,2,3,4,5,6,7,8,9,10,11,12,13,14,15);
}

// B fragment: lane l holds column N=n0+(l&15); halves 0..7 = K in {lo..lo+7},
// halves 8..15 = K in {16+lo..16+lo+7}, lo = (l<16)?0:8
__device__ inline v16h load_bfrag(const half_t* base, int ld, int k0, int n0) {
  const int l = threadIdx.x & 31;
  const int N  = n0 + (l & 15);
  const int lo = (l < 16) ? 0 : 8;
  v16h b;
#pragma unroll
  for (int s = 0; s < 16; ++s) {
    int k = k0 + ((s < 8) ? (lo + s) : (16 + lo + (s - 8)));
    b[s] = base[(size_t)k * ld + N];
  }
  return b;
}

// ordered-float encoding for atomicMax on unsigned
__device__ inline unsigned enc_f(float f) {
  unsigned b = __float_as_uint(f);
  return (b & 0x80000000u) ? ~b : (b | 0x80000000u);
}
__device__ inline float dec_f(unsigned e) {
  unsigned b = (e & 0x80000000u) ? (e & 0x7fffffffu) : ~e;
  return __uint_as_float(b);
}
__device__ inline float sigmoidf_(float x) { return 1.f / (1.f + __expf(-x)); }

// ---------------- weight prep ----------------
__global__ void k_prep_projW(const float* __restrict__ W, half_t* __restrict__ o16) {
  int i = blockIdx.x * 256 + threadIdx.x;             // 128*64 (K padded 110->128)
  if (i >= 128 * DH_) return;
  int k = i / DH_, o = i % DH_;
  o16[i] = (k < DN_) ? (half_t)W[k * DH_ + o] : (half_t)0.0f;
}
__global__ void k_prep_wbig(const float* __restrict__ bW, const float* __restrict__ bb,
                            half_t* __restrict__ o16) {
  int i = blockIdx.x * 256 + threadIdx.x;             // 17*4096
  if (i >= KEF_ * DH_ * DH_) return;
  int k = i / (DH_ * DH_), r = i % (DH_ * DH_);
  o16[i] = (half_t)((k < 16) ? bW[k * DH_ * DH_ + r] : bb[r]);
}
__global__ void k_prep_transpose(const float* __restrict__ W, half_t* __restrict__ o16,
                                 int rows, int cols) {   // W[rows][cols] -> o16[cols][rows]
  int i = blockIdx.x * 256 + threadIdx.x;
  if (i >= rows * cols) return;
  int r = i / cols, c = i % cols;
  o16[c * rows + r] = (half_t)W[i];
}
__global__ void k_cvt_f16(const float* __restrict__ s, half_t* __restrict__ d, int n) {
  int i = blockIdx.x * 256 + threadIdx.x;
  if (i < n) d[i] = (half_t)s[i];
}
__global__ void k_zero_f(float* p, int n) { int i = blockIdx.x*256+threadIdx.x; if (i<n) p[i]=0.f; }
__global__ void k_zero_u(unsigned* p, int n){ int i = blockIdx.x*256+threadIdx.x; if (i<n) p[i]=0u; }
__global__ void k_agg_init(float* __restrict__ agg, const float* __restrict__ bias) {
  int i = blockIdx.x * 256 + threadIdx.x;
  if (i < N_NODES * DH_) agg[i] = bias[i & (DH_ - 1)];
}

// ---------------- node feature projection: x = relu(NA @ projW + b) ----------------
__global__ __launch_bounds__(256)
void k_proj(const float* __restrict__ na, const half_t* __restrict__ pW16,
            const float* __restrict__ pb, float* __restrict__ x, float* __restrict__ x0) {
  const int wave = threadIdx.x >> 5, lane = threadIdx.x & 31;
  const int tile = blockIdx.x * 8 + wave;               // 16-node tile
  if (tile >= N_NODES / 16) return;
  const int row0 = tile * 16;
  const int M  = lane & 15;
  const int lo = (lane < 16) ? 0 : 8;
  const float* row = na + (size_t)(row0 + M) * DN_;
  v16h afr[4];
#pragma unroll
  for (int kc = 0; kc < 4; ++kc) {
    v16h a = {};
#pragma unroll
    for (int h = 0; h < 16; ++h) {
      int k = kc * 32 + ((h < 8) ? (lo + h) : (16 + lo + (h - 8)));
      a[h] = (k < DN_) ? (half_t)row[k] : (half_t)0.0f;
    }
    afr[kc] = a;
  }
#pragma unroll
  for (int nc = 0; nc < 4; ++nc) {
    v8f c = {};
#pragma unroll
    for (int kc = 0; kc < 4; ++kc)
      c = wmma_f32_16x16x32(afr[kc], load_bfrag(pW16, DH_, kc * 32, nc * 16), c);
    const int col = nc * 16 + (lane & 15);
    const int mb  = (lane < 16) ? 0 : 8;
    const float b = pb[col];
#pragma unroll
    for (int j = 0; j < 8; ++j) {
      float v = c[j] + b;
      v = v > 0.f ? v : 0.f;
      size_t idx = (size_t)(row0 + mb + j) * DH_ + col;
      x[idx]  = v;
      x0[idx] = v;
    }
  }
}

// ---------------- NNConv message pass (factorized) ----------------
// msg[e,:] = sum_k efx[e,k] * (x[src[e],:] @ Wbig[k])  -> atomicAdd into agg[dst[e],:]
__global__ __launch_bounds__(128)
void k_nnconv(const float* __restrict__ x, const int* __restrict__ src, const int* __restrict__ dst,
              const float* __restrict__ ea, const float* __restrict__ el,
              const float* __restrict__ centers, const float* __restrict__ beta,
              const half_t* __restrict__ wbig, float* __restrict__ agg) {
  __shared__ alignas(32) half_t xs[16 * 64];   // gathered src features, f16
  __shared__ float efs[16 * KEF_];
  __shared__ int   dsts[16];
  const int t  = threadIdx.x;
  const int e0 = blockIdx.x * 16;
  {
    int r = t >> 3, c0 = (t & 7) * 8;
    const float* xr = x + (size_t)src[e0 + r] * DH_ + c0;
    v8h tv = {};
#pragma unroll
    for (int u = 0; u < 8; ++u) tv[u] = (half_t)xr[u];
    *(v8h*)&xs[r * 64 + c0] = tv;
  }
  for (int i = t; i < 16 * KEF_; i += 128) {
    int r = i / KEF_, k = i % KEF_;
    float v;
    if (k < DE_) v = ea[(e0 + r) * DE_ + k];
    else if (k < DE_ + DL_) {
      float d = el[e0 + r] - centers[k - DE_];
      v = __expf(-beta[k - DE_] * d * d);
    } else v = 1.0f;                      // bias column (bond_b)
    efs[i] = v;
  }
  if (t < 16) dsts[t] = dst[e0 + t];
  __syncthreads();

  const int wave = t >> 5, lane = t & 31;
  const int n0 = wave * 16;               // each wave: 16 edges x 16 outputs
  const v16h a0 = load_afrag(xs, 64, 0, 0);
  const v16h a1 = load_afrag(xs, 64, 0, 32);
  const int mb = (lane < 16) ? 0 : 8;
  float macc[8] = {0.f,0.f,0.f,0.f,0.f,0.f,0.f,0.f};
  for (int k = 0; k < KEF_; ++k) {        // Wbig (139KB f16) is L2-resident
    const half_t* wk = wbig + (size_t)k * DH_ * DH_;
    v8f c = {};
    c = wmma_f32_16x16x32(a0, load_bfrag(wk, DH_, 0,  n0), c);
    c = wmma_f32_16x16x32(a1, load_bfrag(wk, DH_, 32, n0), c);
#pragma unroll
    for (int j = 0; j < 8; ++j) macc[j] += efs[(mb + j) * KEF_ + k] * c[j];
  }
  const int col = n0 + (lane & 15);
#pragma unroll
  for (int j = 0; j < 8; ++j)
    atomicAdd(&agg[(size_t)dsts[mb + j] * DH_ + col], macc[j]);
}

// ---------------- fused GRU cell: gi/gh GEMMs via WMMA + gate epilogue ----------------
__global__ __launch_bounds__(128)
void k_gru(const float* __restrict__ agg, float* __restrict__ x,
           const half_t* __restrict__ WihT, const half_t* __restrict__ WhhT,
           const float* __restrict__ bih, const float* __restrict__ bhh) {
  __shared__ alignas(32) half_t ah[16 * 64];
  __shared__ alignas(32) half_t hh[16 * 64];
  __shared__ float hf[16 * 64];
  __shared__ float gi[16 * 192];
  __shared__ float gh[16 * 192];
  const int t = threadIdx.x;
  const int row0 = blockIdx.x * 16;
  {
    int r = t >> 3, c0 = (t & 7) * 8;
    const float* ar = agg + (size_t)(row0 + r) * DH_ + c0;
    const float* hr = x   + (size_t)(row0 + r) * DH_ + c0;
    v8h ta = {}, th = {};
#pragma unroll
    for (int u = 0; u < 8; ++u) {
      float av = ar[u]; av = av > 0.f ? av : 0.f;   // relu(agg)
      ta[u] = (half_t)av;
      float hv = hr[u];
      th[u] = (half_t)hv;
      hf[r * 64 + c0 + u] = hv;
    }
    *(v8h*)&ah[r * 64 + c0] = ta;
    *(v8h*)&hh[r * 64 + c0] = th;
  }
  __syncthreads();
  const int wave = t >> 5, lane = t & 31;
  const v16h aa0 = load_afrag(ah, 64, 0, 0);
  const v16h aa1 = load_afrag(ah, 64, 0, 32);
  const v16h ha0 = load_afrag(hh, 64, 0, 0);
  const v16h ha1 = load_afrag(hh, 64, 0, 32);
  const int mb = (lane < 16) ? 0 : 8;
#pragma unroll
  for (int q = 0; q < 3; ++q) {                    // 12 n-chunks over 4 waves
    const int n0 = (wave * 3 + q) * 16;
    v8f ci = {}, ch = {};
    ci = wmma_f32_16x16x32(aa0, load_bfrag(WihT, 192, 0,  n0), ci);
    ci = wmma_f32_16x16x32(aa1, load_bfrag(WihT, 192, 32, n0), ci);
    ch = wmma_f32_16x16x32(ha0, load_bfrag(WhhT, 192, 0,  n0), ch);
    ch = wmma_f32_16x16x32(ha1, load_bfrag(WhhT, 192, 32, n0), ch);
    const int col = n0 + (lane & 15);
#pragma unroll
    for (int j = 0; j < 8; ++j) {
      gi[(mb + j) * 192 + col] = ci[j];
      gh[(mb + j) * 192 + col] = ch[j];
    }
  }
  __syncthreads();
#pragma unroll
  for (int u = 0; u < 8; ++u) {
    int lin = t + u * 128;                          // 16x64 elements
    int m = lin >> 6, o = lin & 63;
    float ir = gi[m*192 + o]        + bih[o];
    float iz = gi[m*192 + 64 + o]   + bih[64 + o];
    float in_= gi[m*192 + 128 + o]  + bih[128 + o];
    float hr = gh[m*192 + o]        + bhh[o];
    float hz = gh[m*192 + 64 + o]   + bhh[64 + o];
    float hn = gh[m*192 + 128 + o]  + bhh[128 + o];
    float r  = sigmoidf_(ir + hr);
    float z  = sigmoidf_(iz + hz);
    float nn = tanhf(in_ + r * hn);
    x[(size_t)(row0 + m) * DH_ + o] = (1.f - z) * nn + z * hf[m * 64 + o];
  }
}

// ---------------- Set2Set: fused LSTM layer (B=256 rows, tiny) ----------------
__global__ __launch_bounds__(512)
void k_lstm(const float* __restrict__ inp, int kin,
            const float* __restrict__ Wih, const float* __restrict__ Whh,
            const float* __restrict__ bih, const float* __restrict__ bhh,
            float* __restrict__ hbuf, float* __restrict__ cbuf) {
  __shared__ float si[HQ_];
  __shared__ float sh[H2_];
  __shared__ float g[4 * H2_];
  const int b = blockIdx.x, t = threadIdx.x;
  if (t < kin) si[t] = inp[(size_t)b * kin + t];
  if (t < H2_) sh[t] = hbuf[(size_t)b * H2_ + t];
  __syncthreads();
  float acc = bih[t] + bhh[t];
  const float* wi = Wih + (size_t)t * kin;
  for (int k = 0; k < kin; ++k) acc += si[k] * wi[k];
  const float* wh = Whh + (size_t)t * H2_;
  for (int k = 0; k < H2_; ++k) acc += sh[k] * wh[k];
  g[t] = acc;
  __syncthreads();
  if (t < H2_) {
    float i  = g[t], f = g[H2_ + t], gg = g[2 * H2_ + t], o = g[3 * H2_ + t];
    float c2 = sigmoidf_(f) * cbuf[(size_t)b * H2_ + t] + sigmoidf_(i) * tanhf(gg);
    float h2 = sigmoidf_(o) * tanhf(c2);
    cbuf[(size_t)b * H2_ + t] = c2;
    hbuf[(size_t)b * H2_ + t] = h2;
  }
}

// ---------------- attention: e = feat . q[gid], segment max via ordered-uint ----------------
__global__ __launch_bounds__(256)
void k_attn_e(const float* __restrict__ x, const float* __restrict__ x0,
              const float* __restrict__ q, const int* __restrict__ gid,
              float* __restrict__ earr, unsigned* __restrict__ menc) {
  const int wave = threadIdx.x >> 5, lane = threadIdx.x & 31;
  const int n = blockIdx.x * 8 + wave;
  if (n >= N_NODES) return;
  const int g = gid[n];
  const float* qr = q + (size_t)g * H2_;
  float p = x [(size_t)n*DH_ + lane]      * qr[lane]
          + x [(size_t)n*DH_ + 32 + lane] * qr[32 + lane]
          + x0[(size_t)n*DH_ + lane]      * qr[64 + lane]
          + x0[(size_t)n*DH_ + 32 + lane] * qr[96 + lane];
  for (int off = 16; off > 0; off >>= 1) p += __shfl_xor(p, off, 32);
  if (lane == 0) {
    earr[n] = p;
    atomicMax(&menc[g], enc_f(p));
  }
}
__global__ void k_attn_exp(const int* __restrict__ gid, const unsigned* __restrict__ menc,
                           float* __restrict__ earr, float* __restrict__ ssum) {
  int n = blockIdx.x * 256 + threadIdx.x;
  if (n >= N_NODES) return;
  int g = gid[n];
  float ex = __expf(earr[n] - dec_f(menc[g]));
  earr[n] = ex;
  atomicAdd(&ssum[g], ex);
}
__global__ __launch_bounds__(256)
void k_attn_readout(const float* __restrict__ x, const float* __restrict__ x0,
                    const int* __restrict__ gid, const float* __restrict__ earr,
                    const float* __restrict__ ssum, float* __restrict__ ro) {
  const int wave = threadIdx.x >> 5, lane = threadIdx.x & 31;
  const int n = blockIdx.x * 8 + wave;
  if (n >= N_NODES) return;
  const int g = gid[n];
  float alpha = earr[n] / ssum[g];
  float* r = ro + (size_t)g * H2_;
  atomicAdd(&r[lane],      alpha * x [(size_t)n*DH_ + lane]);
  atomicAdd(&r[32 + lane], alpha * x [(size_t)n*DH_ + 32 + lane]);
  atomicAdd(&r[64 + lane], alpha * x0[(size_t)n*DH_ + lane]);
  atomicAdd(&r[96 + lane], alpha * x0[(size_t)n*DH_ + 32 + lane]);
}
__global__ void k_qstar(const float* __restrict__ hl1, const float* __restrict__ ro,
                        float* __restrict__ qstar) {
  int i = blockIdx.x * 256 + threadIdx.x;
  if (i >= BGRAPH * HQ_) return;
  int b = i >> 8, d = i & 255;
  qstar[i] = (d < H2_) ? hl1[b * H2_ + d] : ro[b * H2_ + (d - H2_)];
}

// ---------------- final: out = PReLU(q_star @ sparsify_W + b) via WMMA ----------------
__global__ __launch_bounds__(128)
void k_sparsify(const half_t* __restrict__ qs16, const half_t* __restrict__ sW16,
                const float* __restrict__ sb, const float* __restrict__ pa,
                float* __restrict__ out) {
  const int wave = threadIdx.x >> 5, lane = threadIdx.x & 31;
  const int wg = blockIdx.x * 4 + wave;     // 4096 waves: 16 m-tiles x 256 n-chunks
  const int mt = wg >> 8, nc = wg & 255;
  const int n0 = nc * 16;
  v8f c = {};
#pragma unroll
  for (int kc = 0; kc < 8; ++kc) {
    v16h a = load_afrag(qs16, HQ_, mt * 16, kc * 32);
    v16h b = load_bfrag(sW16, DHID_, kc * 32, n0);
    c = wmma_f32_16x16x32(a, b, c);
  }
  const float slope = pa[0];
  const int col = n0 + (lane & 15);
  const int mb  = (lane < 16) ? 0 : 8;
  const float bias = sb[col];
#pragma unroll
  for (int j = 0; j < 8; ++j) {
    float v = c[j] + bias;
    v = (v >= 0.f) ? v : slope * v;
    out[(size_t)(mt * 16 + mb + j) * DHID_ + col] = v;
  }
}

// ---------------- host launcher ----------------
extern "C" void kernel_launch(void* const* d_in, const int* in_sizes, int n_in,
                              void* d_out, int out_size, void* d_ws, size_t ws_size,
                              hipStream_t stream) {
  (void)in_sizes; (void)n_in; (void)out_size; (void)ws_size;
  const float* node_attr = (const float*)d_in[0];
  const float* edge_attr = (const float*)d_in[1];
  const float* edge_len  = (const float*)d_in[2];
  const int*   src       = (const int*)d_in[3];
  const int*   dst       = (const int*)d_in[4];
  const int*   gid       = (const int*)d_in[5];
  const float* projW     = (const float*)d_in[6];
  const float* projb     = (const float*)d_in[7];
  const float* centers   = (const float*)d_in[8];
  const float* beta      = (const float*)d_in[9];
  const float* bondW     = (const float*)d_in[10];
  const float* bondb     = (const float*)d_in[11];
  const float* nbias     = (const float*)d_in[12];
  const float* gWih      = (const float*)d_in[13];
  const float* gWhh      = (const float*)d_in[14];
  const float* gbih      = (const float*)d_in[15];
  const float* gbhh      = (const float*)d_in[16];
  const float* lWih0     = (const float*)d_in[17];
  const float* lWhh0     = (const float*)d_in[18];
  const float* lbih0     = (const float*)d_in[19];
  const float* lbhh0     = (const float*)d_in[20];
  const float* lWih1     = (const float*)d_in[21];
  const float* lWhh1     = (const float*)d_in[22];
  const float* lbih1     = (const float*)d_in[23];
  const float* lbhh1     = (const float*)d_in[24];
  const float* sW        = (const float*)d_in[25];
  const float* sb        = (const float*)d_in[26];
  const float* pa        = (const float*)d_in[27];
  float* out = (float*)d_out;

  char* w = (char*)d_ws;
  auto alloc = [&](size_t bytes) { char* p = w; w += (bytes + 127) & ~(size_t)127; return p; };
  float*    x     = (float*)alloc((size_t)N_NODES * DH_ * 4);
  float*    x0    = (float*)alloc((size_t)N_NODES * DH_ * 4);
  float*    agg   = (float*)alloc((size_t)N_NODES * DH_ * 4);
  float*    hl0   = (float*)alloc((size_t)BGRAPH * H2_ * 4);
  float*    cl0   = (float*)alloc((size_t)BGRAPH * H2_ * 4);
  float*    hl1   = (float*)alloc((size_t)BGRAPH * H2_ * 4);
  float*    cl1   = (float*)alloc((size_t)BGRAPH * H2_ * 4);
  float*    qstar = (float*)alloc((size_t)BGRAPH * HQ_ * 4);
  float*    earr  = (float*)alloc((size_t)N_NODES * 4);
  float*    ssum  = (float*)alloc((size_t)BGRAPH * 4);
  float*    ro    = (float*)alloc((size_t)BGRAPH * H2_ * 4);
  unsigned* menc  = (unsigned*)alloc((size_t)BGRAPH * 4);
  half_t*   pW16  = (half_t*)alloc((size_t)128 * DH_ * 2);
  half_t*   wbig16= (half_t*)alloc((size_t)KEF_ * DH_ * DH_ * 2);
  half_t*   WihT16= (half_t*)alloc((size_t)DH_ * 192 * 2);
  half_t*   WhhT16= (half_t*)alloc((size_t)DH_ * 192 * 2);
  half_t*   sW16  = (half_t*)alloc((size_t)HQ_ * DHID_ * 2);
  half_t*   qs16  = (half_t*)alloc((size_t)BGRAPH * HQ_ * 2);

  auto cdiv = [](int a, int b) { return (a + b - 1) / b; };

  // weight prep (deterministic, each launch)
  k_prep_projW<<<cdiv(128 * DH_, 256), 256, 0, stream>>>(projW, pW16);
  k_prep_wbig<<<cdiv(KEF_ * DH_ * DH_, 256), 256, 0, stream>>>(bondW, bondb, wbig16);
  k_prep_transpose<<<cdiv(192 * DH_, 256), 256, 0, stream>>>(gWih, WihT16, 192, DH_);
  k_prep_transpose<<<cdiv(192 * DH_, 256), 256, 0, stream>>>(gWhh, WhhT16, 192, DH_);
  k_cvt_f16<<<cdiv(HQ_ * DHID_, 256), 256, 0, stream>>>(sW, sW16, HQ_ * DHID_);

  // node projection
  k_proj<<<cdiv(N_NODES / 16, 8), 256, 0, stream>>>(node_attr, pW16, projb, x, x0);

  // Set2Set state init
  k_zero_f<<<cdiv(BGRAPH * H2_, 256), 256, 0, stream>>>(hl0, BGRAPH * H2_);
  k_zero_f<<<cdiv(BGRAPH * H2_, 256), 256, 0, stream>>>(cl0, BGRAPH * H2_);
  k_zero_f<<<cdiv(BGRAPH * H2_, 256), 256, 0, stream>>>(hl1, BGRAPH * H2_);
  k_zero_f<<<cdiv(BGRAPH * H2_, 256), 256, 0, stream>>>(cl1, BGRAPH * H2_);
  k_zero_f<<<cdiv(BGRAPH * HQ_, 256), 256, 0, stream>>>(qstar, BGRAPH * HQ_);

  // message passing steps
  for (int s = 0; s < STEPS_; ++s) {
    k_agg_init<<<cdiv(N_NODES * DH_, 256), 256, 0, stream>>>(agg, nbias);
    k_nnconv<<<N_EDGES / 16, 128, 0, stream>>>(x, src, dst, edge_attr, edge_len,
                                               centers, beta, wbig16, agg);
    k_gru<<<N_NODES / 16, 128, 0, stream>>>(agg, x, WihT16, WhhT16, gbih, gbhh);
  }

  // Set2Set pooling
  for (int it = 0; it < POOL_IT; ++it) {
    k_lstm<<<BGRAPH, 512, 0, stream>>>(qstar, HQ_, lWih0, lWhh0, lbih0, lbhh0, hl0, cl0);
    k_lstm<<<BGRAPH, 512, 0, stream>>>(hl0,   H2_, lWih1, lWhh1, lbih1, lbhh1, hl1, cl1);
    k_zero_u<<<1, 256, 0, stream>>>(menc, BGRAPH);
    k_zero_f<<<1, 256, 0, stream>>>(ssum, BGRAPH);
    k_zero_f<<<cdiv(BGRAPH * H2_, 256), 256, 0, stream>>>(ro, BGRAPH * H2_);
    k_attn_e<<<N_NODES / 8, 256, 0, stream>>>(x, x0, hl1, gid, earr, menc);
    k_attn_exp<<<cdiv(N_NODES, 256), 256, 0, stream>>>(gid, menc, earr, ssum);
    k_attn_readout<<<N_NODES / 8, 256, 0, stream>>>(x, x0, gid, earr, ssum, ro);
    k_qstar<<<cdiv(BGRAPH * HQ_, 256), 256, 0, stream>>>(hl1, ro, qstar);
  }

  // final linear + PReLU
  k_cvt_f16<<<cdiv(BGRAPH * HQ_, 256), 256, 0, stream>>>(qstar, qs16, BGRAPH * HQ_);
  k_sparsify<<<1024, 128, 0, stream>>>(qs16, sW16, sb, pa, out);
}
